// NO1_13288628814181
// MI455X (gfx1250) — compile-verified, matrix operand order checked by reading, not run on hardware
//
#include <hip/hip_runtime.h>

#define HH 160
#define WWD 100
#define HID 32
#define BB 32
#define NPIX (BB*HH*WWD)          // 512000
#define ROWS (BB*HID*HH)          // 163840
#define PI2 6.28318530717958647692f

typedef __attribute__((ext_vector_type(16))) _Float16 v16h;
typedef __attribute__((ext_vector_type(8)))  float    v8f;

__device__ __forceinline__ float gelu_f(float x){
  return 0.5f*x*(1.0f + erff(x*0.7071067811865475f));
}
__device__ __forceinline__ int frag_k0(int v, int half){
  return (v < 4 ? 2*v : 16 + 2*(v-4)) + 8*half;
}
__device__ __forceinline__ v8f wmma16(v16h a, v16h b, v8f c){
  return __builtin_amdgcn_wmma_f32_16x16x32_f16(false, a, false, b, (short)0, c, false, false);
}

// ---------------- one-time f32 -> f16 weight conversion ----------------
__global__ void k_cvt(const float* __restrict__ src, _Float16* __restrict__ dst, int n){
  int t = blockIdx.x*blockDim.x + threadIdx.x;
  if (t < n) dst[t] = (_Float16)src[t];
}

// ---------------- one-time twiddle tables (f16, fragment-ready) ----------------
// forward W-DFT B table: tw[k*16+n], k<128, n<8: cos(2pi k n/100); n in 8..15: -sin(...)
__global__ void k_tw_dftw(_Float16* __restrict__ tw){
  int t = blockIdx.x*blockDim.x + threadIdx.x;
  if (t >= 128*16) return;
  int k = t >> 4, n = t & 15;
  int m = (n < 8) ? n : n - 8;
  float ang = PI2 * (float)((k*m)%100) * 0.01f;
  tw[t] = (_Float16)((n < 8) ? __cosf(ang) : -__sinf(ang));
}
// inverse W-DFT B table: itab[k*112+w], k<32, w<112 (irfft2 1/(H*W) folded in)
__global__ void k_tw_idftw(_Float16* __restrict__ itab){
  int t = blockIdx.x*blockDim.x + threadIdx.x;
  if (t >= 32*112) return;
  int k = t / 112, w = t % 112;
  const float inv = 1.0f/16000.0f;
  float val;
  if (k == 0)       val = inv;
  else if (k < 8)   val =  2.0f*inv*__cosf(PI2*(float)((k*w)%100)*0.01f);
  else if (k < 16)  val = -2.0f*inv*__sinf(PI2*(float)(((k-8)*w)%100)*0.01f);
  else              val = 0.0f;
  itab[t] = (_Float16)val;
}

// ---------------- Tucker reconstruction (tiny, VALU) ----------------
__global__ void k_tucker1(const float* cr, const float* ci,
                          const float* fr, const float* fi, float2* T){
  int t = blockIdx.x*blockDim.x + threadIdx.x;
  const int N = 8*27*27*7*7;
  if (t >= N) return;
  int e = t%7, d = (t/7)%7, c = (t/49)%27, b = (t/(49*27))%27, x = t/(49*27*27);
  float aR=0.f, aI=0.f;
  for (int a=0;a<7;a++){
    int ix = (((a*27+b)*27+c)*7+d)*7+e;
    float tr=cr[ix], ti=ci[ix], gr=fr[x*7+a], gi=fi[x*7+a];
    aR += tr*gr - ti*gi;  aI += tr*gi + ti*gr;
  }
  T[t] = make_float2(aR,aI);
}
__global__ void k_tucker2(const float2* P, const float* fr, const float* fi, float2* T){
  int t = blockIdx.x*blockDim.x + threadIdx.x;
  const int N = 8*32*27*7*7;
  if (t >= N) return;
  int e=t%7, d=(t/7)%7, c=(t/49)%27, y=(t/(49*27))%32, x=t/(49*27*32);
  float aR=0.f, aI=0.f;
  for (int b=0;b<27;b++){
    float2 p = P[(((x*27+b)*27+c)*7+d)*7+e];
    float gr=fr[y*27+b], gi=fi[y*27+b];
    aR += p.x*gr - p.y*gi;  aI += p.x*gi + p.y*gr;
  }
  T[t] = make_float2(aR,aI);
}
__global__ void k_tucker3(const float2* P, const float* fr, const float* fi, float2* T){
  int t = blockIdx.x*blockDim.x + threadIdx.x;
  const int N = 8*32*32*7*7;
  if (t >= N) return;
  int e=t%7, d=(t/7)%7, z=(t/49)%32, y=(t/(49*32))%32, x=t/(49*32*32);
  float aR=0.f, aI=0.f;
  for (int c=0;c<27;c++){
    float2 p = P[(((x*32+y)*27+c)*7+d)*7+e];
    float gr=fr[z*27+c], gi=fi[z*27+c];
    aR += p.x*gr - p.y*gi;  aI += p.x*gi + p.y*gr;
  }
  T[t] = make_float2(aR,aI);
}
__global__ void k_tucker4(const float2* P, const float* fr, const float* fi, float2* T){
  int t = blockIdx.x*blockDim.x + threadIdx.x;
  const int N = 8*32*32*8*7;
  if (t >= N) return;
  int e=t%7, u=(t/7)%8, z=(t/56)%32, y=(t/(56*32))%32, x=t/(56*32*32);
  float aR=0.f, aI=0.f;
  for (int d=0;d<7;d++){
    float2 p = P[(((x*32+y)*32+z)*7+d)*7+e];
    float gr=fr[u*7+d], gi=fi[u*7+d];
    aR += p.x*gr - p.y*gi;  aI += p.x*gi + p.y*gr;
  }
  T[t] = make_float2(aR,aI);
}
__global__ void k_tucker5(const float2* P, const float* fr, const float* fi, float2* Wt){
  int t = blockIdx.x*blockDim.x + threadIdx.x;
  const int N = 8*32*32*8*8;
  if (t >= N) return;
  int v=t%8, u=(t/8)%8, z=(t/64)%32, y=(t/2048)%32, x=t/65536;
  float aR=0.f, aI=0.f;
  for (int e=0;e<7;e++){
    float2 p = P[(((x*32+y)*32+z)*8+u)*7+e];
    float gr=fr[v*7+e], gi=fi[v*7+e];
    aR += p.x*gr - p.y*gi;  aI += p.x*gi + p.y*gr;
  }
  Wt[t] = make_float2(aR,aI);
}

// ---------------- Preproc GEMM [32x16000x256] + GELU (WMMA) ----------------
__global__ void k_preproc(const float* __restrict__ x, const _Float16* __restrict__ lw16,
                          const float* __restrict__ lin_b, float* __restrict__ h0){
  int wid  = (blockIdx.x*blockDim.x + threadIdx.x) >> 5;
  int lane = threadIdx.x & 31;
  if (wid >= 2*1000) return;
  int mt = wid/1000, nt = wid%1000;
  int half = lane>>4, l16 = lane&15;
  int rowA = mt*16 + l16, colB = nt*16 + l16;
  v8f acc = {0.f,0.f,0.f,0.f,0.f,0.f,0.f,0.f};
  for (int kc=0; kc<8; ++kc){
    v16h a, b;
#pragma unroll
    for (int v=0; v<8; ++v){
      int k = kc*32 + frag_k0(v, half);
      a[2*v]   = (_Float16)x[rowA*256 + k];
      a[2*v+1] = (_Float16)x[rowA*256 + k + 1];
      b[2*v]   = lw16[colB*256 + k];
      b[2*v+1] = lw16[colB*256 + k + 1];
    }
    acc = wmma16(a, b, acc);
  }
#pragma unroll
  for (int v=0; v<8; ++v){
    int row = mt*16 + v + 8*half, col = nt*16 + l16;
    h0[row*16000 + col] = gelu_f(acc[v] + lin_b[col]);
  }
}

// ---------------- Fused lifting MLP: gelu(h0*w1+b1) @ w2^T + b2 (WMMA) ----------------
__global__ void k_lift(const float* __restrict__ h0, const float* __restrict__ w1,
                       const float* __restrict__ b1, const _Float16* __restrict__ w2_16,
                       const float* __restrict__ b2, _Float16* __restrict__ h16){
  int wid  = (blockIdx.x*blockDim.x + threadIdx.x) >> 5;
  int lane = threadIdx.x & 31;
  if (wid >= NPIX/16) return;
  int half = lane>>4, l16 = lane&15;
  int pix0 = wid*16;
  float hv = h0[pix0 + l16];
  v8f c0 = {0.f,0.f,0.f,0.f,0.f,0.f,0.f,0.f};
  v8f c1 = {0.f,0.f,0.f,0.f,0.f,0.f,0.f,0.f};
  for (int kc=0; kc<8; ++kc){
    v16h a, bA, bB;
#pragma unroll
    for (int v=0; v<8; ++v){
      int k = kc*32 + frag_k0(v, half);
      a[2*v]   = (_Float16)gelu_f(hv*w1[k]   + b1[k]);
      a[2*v+1] = (_Float16)gelu_f(hv*w1[k+1] + b1[k+1]);
      bA[2*v]   = w2_16[l16*256 + k];
      bA[2*v+1] = w2_16[l16*256 + k + 1];
      bB[2*v]   = w2_16[(l16+16)*256 + k];
      bB[2*v+1] = w2_16[(l16+16)*256 + k + 1];
    }
    c0 = wmma16(a, bA, c0);
    c1 = wmma16(a, bB, c1);
  }
#pragma unroll
  for (int v=0; v<8; ++v){
    int pix = pix0 + v + 8*half;
    int b = pix/16000, hw = pix%16000;
    h16[(b*32 + l16     )*16000 + hw] = (_Float16)(c0[v] + b2[l16]);
    h16[(b*32 + l16 + 16)*16000 + hw] = (_Float16)(c1[v] + b2[l16+16]);
  }
}

// ---------------- Per-pixel skip GEMM 32x32 (WMMA) ----------------
__global__ void k_skip(const _Float16* __restrict__ h16, const _Float16* __restrict__ sw16,
                       float* __restrict__ s){
  int wid  = (blockIdx.x*blockDim.x + threadIdx.x) >> 5;
  int lane = threadIdx.x & 31;
  if (wid >= NPIX/16) return;
  int half = lane>>4, l16 = lane&15;
  int pix0 = wid*16;
  int pixA = pix0 + l16, bA = pixA/16000, hwA = pixA%16000;
  v16h a, b0, b1;
#pragma unroll
  for (int v=0; v<8; ++v){
    int k0 = frag_k0(v, half);
    a[2*v]   = h16[(bA*32 + k0    )*16000 + hwA];
    a[2*v+1] = h16[(bA*32 + k0 + 1)*16000 + hwA];
    b0[2*v]   = sw16[l16*32 + k0];
    b0[2*v+1] = sw16[l16*32 + k0 + 1];
    b1[2*v]   = sw16[(l16+16)*32 + k0];
    b1[2*v+1] = sw16[(l16+16)*32 + k0 + 1];
  }
  v8f c0 = {0.f,0.f,0.f,0.f,0.f,0.f,0.f,0.f};
  v8f c1 = {0.f,0.f,0.f,0.f,0.f,0.f,0.f,0.f};
  c0 = wmma16(a, b0, c0);
  c1 = wmma16(a, b1, c1);
#pragma unroll
  for (int v=0; v<8; ++v){
    int pix = pix0 + v + 8*half;
    int b = pix/16000, hw = pix%16000;
    s[(b*32 + l16     )*16000 + hw] = c0[v];
    s[(b*32 + l16 + 16)*16000 + hw] = c1[v];
  }
}

// ---------------- Truncated DFT along W: 100 -> 8 modes (WMMA, K padded to 128) ----------
__global__ void k_dftw(const _Float16* __restrict__ h16, const _Float16* __restrict__ tw,
                       float* __restrict__ Xw){
  int wid  = (blockIdx.x*blockDim.x + threadIdx.x) >> 5;
  int lane = threadIdx.x & 31;
  if (wid >= ROWS/16) return;
  int half = lane>>4, l16 = lane&15;
  int rA = wid*16 + l16;
  v8f acc = {0.f,0.f,0.f,0.f,0.f,0.f,0.f,0.f};
  for (int kc=0; kc<4; ++kc){
    v16h a, b;
#pragma unroll
    for (int v=0; v<8; ++v){
      int k = kc*32 + frag_k0(v, half);
      a[2*v]   = (k   < 100) ? h16[rA*100 + k]     : (_Float16)0.f;
      a[2*v+1] = (k+1 < 100) ? h16[rA*100 + k + 1] : (_Float16)0.f;
      b[2*v]   = tw[k*16 + l16];
      b[2*v+1] = tw[(k+1)*16 + l16];
    }
    acc = wmma16(a, b, acc);
  }
#pragma unroll
  for (int v=0; v<8; ++v){
    int r2 = wid*16 + v + 8*half;
    Xw[r2*16 + l16] = acc[v];
  }
}

// ---------------- DFT along H for 16 kept modes (VALU, tiny) ----------------
__global__ void k_dfth(const float* __restrict__ Xw, float2* __restrict__ Xf){
  int t = blockIdx.x*blockDim.x + threadIdx.x;
  if (t >= 32*32*16*8) return;
  int m2 = t&7, m1 = (t>>3)&15, bc = t>>7;
  int kf = (m1 < 8) ? m1 : 144 + m1;
  float aR=0.f, aI=0.f;
  for (int hh=0; hh<160; ++hh){
    int base = (bc*160 + hh)*16;
    float xr = Xw[base + m2], xi = Xw[base + 8 + m2];
    float ang = PI2 * (float)((hh*kf)%160) * 0.00625f;
    float sn, cs; __sincosf(ang, &sn, &cs);
    aR += xr*cs + xi*sn;          // * exp(-i ang)
    aI += xi*cs - xr*sn;
  }
  Xf[t] = make_float2(aR, aI);
}

// ---------------- Spectral complex channel mix (VALU, tiny) ----------------
__global__ void k_specmul(const float2* __restrict__ Xf, const float2* __restrict__ Wt,
                          float2* __restrict__ Y, int l){
  int t = blockIdx.x*blockDim.x + threadIdx.x;
  if (t >= 32*32*16*8) return;
  int m2=t&7, m1=(t>>3)&15, o=(t>>7)&31, b=t>>12;
  int sel = 2*l + (m1>=8 ? 1 : 0), m1m = m1&7;
  float aR=0.f, aI=0.f;
  for (int i=0;i<32;i++){
    float2 xf = Xf[((b*32+i)*16 + m1)*8 + m2];
    float2 w  = Wt[(((sel*32+i)*32 + o)*8 + m1m)*8 + m2];
    aR += xf.x*w.x - xf.y*w.y;
    aI += xf.x*w.y + xf.y*w.x;
  }
  Y[t] = make_float2(aR, aI);
}

// ---------------- Inverse DFT along H (VALU), outputs f16 rows for WMMA A --------------
__global__ void k_idfth(const float2* __restrict__ Y, _Float16* __restrict__ Zh){
  int t = blockIdx.x*blockDim.x + threadIdx.x;
  if (t >= 32*32*160*8) return;
  int m2 = t&7, hh = (t>>3)%160, bo = t/(160*8);
  float aR=0.f, aI=0.f;
#pragma unroll
  for (int m1=0; m1<16; ++m1){
    int kf = (m1 < 8) ? m1 : 144 + m1;
    float ang = PI2 * (float)((hh*kf)%160) * 0.00625f;
    float sn, cs; __sincosf(ang, &sn, &cs);
    float2 y = Y[(bo*16 + m1)*8 + m2];
    aR += y.x*cs - y.y*sn;        // * exp(+i ang)
    aI += y.x*sn + y.y*cs;
  }
  int r = bo*160 + hh;
  Zh[r*16 + m2]     = (_Float16)aR;
  Zh[r*16 + 8 + m2] = (_Float16)aI;
}

// ---------------- Fused inverse rDFT along W + skip + bias + gelu (WMMA) ---------------
__global__ void k_idftw(const _Float16* __restrict__ Zh, const _Float16* __restrict__ itab,
                        const float* __restrict__ s, const float* __restrict__ sbias,
                        _Float16* __restrict__ h16, int apply_gelu){
  int wid  = (blockIdx.x*blockDim.x + threadIdx.x) >> 5;
  int lane = threadIdx.x & 31;
  if (wid >= (ROWS/16)*7) return;
  int rt = wid/7, nt = wid%7;
  int half = lane>>4, l16 = lane&15;
  int rA = rt*16 + l16;
  int wB = nt*16 + l16;
  v16h a, b;
#pragma unroll
  for (int v=0; v<8; ++v){
    int k0 = frag_k0(v, half);
    if (v < 4){                       // k0 in [0,16): real data
      a[2*v]   = Zh[rA*16 + k0];
      a[2*v+1] = Zh[rA*16 + k0 + 1];
    } else {                          // K padding
      a[2*v] = (_Float16)0.f; a[2*v+1] = (_Float16)0.f;
    }
    b[2*v]   = itab[k0*112 + wB];
    b[2*v+1] = itab[(k0+1)*112 + wB];
  }
  v8f c;
#pragma unroll
  for (int v=0; v<8; ++v){
    int r2 = rt*16 + v + 8*half;
    int wcol = nt*16 + l16;
    int wc = (wcol < 100) ? wcol : 99;
    int o = (r2/160) & 31;
    c[v] = s[r2*100 + wc] + sbias[o];
  }
  c = wmma16(a, b, c);
#pragma unroll
  for (int v=0; v<8; ++v){
    int r2 = rt*16 + v + 8*half;
    int wcol = nt*16 + l16;
    if (wcol < 100){
      float val = c[v];
      if (apply_gelu) val = gelu_f(val);
      h16[r2*100 + wcol] = (_Float16)val;
    }
  }
}

// ---------------- Fused projection MLP 32->64(gelu)->1 (WMMA + shuffle reduce) ---------
__global__ void k_proj(const _Float16* __restrict__ h16, const _Float16* __restrict__ w1_16,
                       const float* __restrict__ b1, const float* __restrict__ w2,
                       const float* __restrict__ b2, float* __restrict__ out){
  int wid  = (blockIdx.x*blockDim.x + threadIdx.x) >> 5;
  int lane = threadIdx.x & 31;
  if (wid >= NPIX/16) return;
  int half = lane>>4, l16 = lane&15;
  int pix0 = wid*16;
  int pixA = pix0 + l16, bA = pixA/16000, hwA = pixA%16000;
  v16h a;
#pragma unroll
  for (int v=0; v<8; ++v){
    int k0 = frag_k0(v, half);
    a[2*v]   = h16[(bA*32 + k0    )*16000 + hwA];
    a[2*v+1] = h16[(bA*32 + k0 + 1)*16000 + hwA];
  }
  float partial[8];
#pragma unroll
  for (int v=0; v<8; ++v) partial[v] = 0.f;
  for (int nt=0; nt<4; ++nt){
    v16h bf;
    int n = nt*16 + l16;
#pragma unroll
    for (int v=0; v<8; ++v){
      int k0 = frag_k0(v, half);
      bf[2*v]   = w1_16[n*32 + k0];
      bf[2*v+1] = w1_16[n*32 + k0 + 1];
    }
    v8f c = {0.f,0.f,0.f,0.f,0.f,0.f,0.f,0.f};
    c = wmma16(a, bf, c);
#pragma unroll
    for (int v=0; v<8; ++v){
      float tval = gelu_f(c[v] + b1[n]);
      partial[v] += tval * w2[n];
    }
  }
#pragma unroll
  for (int v=0; v<8; ++v){
    float p = partial[v];
    p += __shfl_xor(p, 1, 32);
    p += __shfl_xor(p, 2, 32);
    p += __shfl_xor(p, 4, 32);
    p += __shfl_xor(p, 8, 32);
    if (l16 == 0){
      int pix = pix0 + v + 8*half;
      out[pix] = p + b2[0];
    }
  }
}

extern "C" void kernel_launch(void* const* d_in, const int* in_sizes, int n_in,
                              void* d_out, int out_size, void* d_ws, size_t ws_size,
                              hipStream_t stream){
  const float* x     = (const float*)d_in[0];
  const float* lin_w = (const float*)d_in[1];
  const float* lin_b = (const float*)d_in[2];
  const float* lw1   = (const float*)d_in[3];
  const float* lb1   = (const float*)d_in[4];
  const float* lw2   = (const float*)d_in[5];
  const float* lb2   = (const float*)d_in[6];
  const float* skw   = (const float*)d_in[7];
  const float* cre   = (const float*)d_in[8];
  const float* cim   = (const float*)d_in[9];
  const float* f0r   = (const float*)d_in[10];
  const float* f0i   = (const float*)d_in[11];
  const float* f1r   = (const float*)d_in[12];
  const float* f1i   = (const float*)d_in[13];
  const float* f2r   = (const float*)d_in[14];
  const float* f2i   = (const float*)d_in[15];
  const float* f3r   = (const float*)d_in[16];
  const float* f3i   = (const float*)d_in[17];
  const float* f4r   = (const float*)d_in[18];
  const float* f4i   = (const float*)d_in[19];
  const float* sbias = (const float*)d_in[20];
  const float* pw1   = (const float*)d_in[21];
  const float* pb1   = (const float*)d_in[22];
  const float* pw2   = (const float*)d_in[23];
  const float* pb2   = (const float*)d_in[24];
  float* out = (float*)d_out;
  (void)in_sizes; (void)n_in; (void)out_size; (void)ws_size;

  char* base = (char*)d_ws;
  size_t off = 0;
  auto walloc = [&](size_t bytes)->char*{
    char* p = base + off;
    off = (off + bytes + 255) & ~(size_t)255;
    return p;
  };
  float2*   bufA  = (float2*)  walloc((size_t)401408*sizeof(float2));
  float2*   bufB  = (float2*)  walloc((size_t)458752*sizeof(float2));
  float2*   Wt    = (float2*)  walloc((size_t)524288*sizeof(float2));
  float*    h0    = (float*)   walloc((size_t)512000*sizeof(float));
  _Float16* h16   = (_Float16*)walloc((size_t)16384000*sizeof(_Float16));
  float*    s     = (float*)   walloc((size_t)16384000*sizeof(float));
  float*    Xw    = (float*)   walloc((size_t)ROWS*16*sizeof(float));
  float2*   Xf    = (float2*)  walloc((size_t)131072*sizeof(float2));
  float2*   Y     = (float2*)  walloc((size_t)131072*sizeof(float2));
  _Float16* Zh    = (_Float16*)walloc((size_t)ROWS*16*sizeof(_Float16));
  _Float16* lw16  = (_Float16*)walloc((size_t)4096000*sizeof(_Float16)); // lin_w f16
  _Float16* lw2_16= (_Float16*)walloc((size_t)8192*sizeof(_Float16));    // lift_w2 f16
  _Float16* skw16 = (_Float16*)walloc((size_t)4096*sizeof(_Float16));    // skip_w f16 (4 layers)
  _Float16* pw1_16= (_Float16*)walloc((size_t)2048*sizeof(_Float16));    // proj_w1 f16
  _Float16* tw    = (_Float16*)walloc((size_t)128*16*sizeof(_Float16));  // fwd W-DFT table
  _Float16* itab  = (_Float16*)walloc((size_t)32*112*sizeof(_Float16));  // inv W-DFT table

  // One-time weight f16 conversion + twiddle tables
  k_cvt<<<(4096000+255)/256, 256, 0, stream>>>(lin_w, lw16, 4096000);
  k_cvt<<<(8192+255)/256,    256, 0, stream>>>(lw2,   lw2_16, 8192);
  k_cvt<<<(4096+255)/256,    256, 0, stream>>>(skw,   skw16, 4096);
  k_cvt<<<(2048+255)/256,    256, 0, stream>>>(pw1,   pw1_16, 2048);
  k_tw_dftw <<<8,  256, 0, stream>>>(tw);
  k_tw_idftw<<<14, 256, 0, stream>>>(itab);

  // Tucker reconstruction of spectral weights (once, tiny)
  k_tucker1<<<(285768+255)/256, 256, 0, stream>>>(cre, cim, f0r, f0i, bufA);
  k_tucker2<<<(338688+255)/256, 256, 0, stream>>>(bufA, f1r, f1i, bufB);
  k_tucker3<<<(401408+255)/256, 256, 0, stream>>>(bufB, f2r, f2i, bufA);
  k_tucker4<<<(458752+255)/256, 256, 0, stream>>>(bufA, f3r, f3i, bufB);
  k_tucker5<<<(524288+255)/256, 256, 0, stream>>>(bufB, f4r, f4i, Wt);

  // Preproc GEMM + GELU: 2000 waves, 8 waves/block
  k_preproc<<<250, 256, 0, stream>>>(x, lw16, lin_b, h0);
  // Fused lifting MLP: 32768 waves
  k_lift<<<4096, 256, 0, stream>>>(h0, lw1, lb1, lw2_16, lb2, h16);

  for (int l=0; l<4; ++l){
    k_skip   <<<4096, 256, 0, stream>>>(h16, skw16 + l*HID*HID, s);
    k_dftw   <<<1280, 256, 0, stream>>>(h16, tw, Xw);
    k_dfth   <<<512,  256, 0, stream>>>(Xw, Xf);
    k_specmul<<<512,  256, 0, stream>>>(Xf, Wt, Y, l);
    k_idfth  <<<5120, 256, 0, stream>>>(Y, Zh);
    k_idftw  <<<8960, 256, 0, stream>>>(Zh, itab, s, sbias + l*HID, h16, (l<3) ? 1 : 0);
  }
  // Fused projection MLP
  k_proj<<<4096, 256, 0, stream>>>(h16, pw1_16, pb1, pw2, pb2, out);
}
